// Generator_59227599011854
// MI455X (gfx1250) — matrix operand order repaired and checked by reading, not compiled, not run on hardware
//
#include <hip/hip_runtime.h>

// ---------------------------------------------------------------------------
// Problem constants (from reference)
// ---------------------------------------------------------------------------
#define OLDN   3840
#define NTOT   4096
#define NNEW   256
#define LATD   128
#define HIDD   256
#define R1D    256
#define R2D    256
#define BUDGET_A 5000
#define BUDGET_X 1024
#define ALPHA_C  2.0f
#define NEGINF  (-1e20f)

typedef __bf16 bf16_t;
typedef __attribute__((ext_vector_type(16))) __bf16 v16bf;
typedef __attribute__((ext_vector_type(8)))  float  v8f;

// ---------------------------------------------------------------------------
// WMMA helpers (CDNA5 gfx1250, wave32): D = A(16x32 bf16) x B(32x16 bf16) + C
// A-layout as loaded here (16x32): lanes 0-15 -> M=lane, elems 0..15 = K 0..15;
//   lanes 16-31 -> M=lane-16, elems = K 16..31 (same k order as B).
// B-layout as staged here (32x16): lanes 0-15 -> N=lane, elems = K 0..15;
//   lanes 16-31 -> N=lane-16, elems = K 16..31.
// C/D-layout: vgpr r, lane l -> (M = r + 8*(l>>4), N = l&15).
// ---------------------------------------------------------------------------
__device__ __forceinline__ v8f wmma_bf16(v16bf a, v16bf b, v8f c) {
  return __builtin_amdgcn_wmma_f32_16x16x32_bf16(false, a, false, b,
                                                 (short)0, c, false, false);
}

// A fragment from a row-major bf16 LDS tile (stride in elements).
__device__ __forceinline__ v16bf lds_load_a16x32(const bf16_t* base, int stride) {
  const int lane = threadIdx.x & 31;
  const int m    = lane & 15;
  const int khi  = lane >> 4;
  const bf16_t* row = base + m * stride;
  v16bf a;
#pragma unroll
  for (int e = 0; e < 8; ++e) a[e]     = row[khi * 16 + e];     // 16B chunk
#pragma unroll
  for (int e = 0; e < 8; ++e) a[e + 8] = row[khi * 16 + 8 + e]; // 16B chunk
  return a;
}

// B fragment from a PRE-SWIZZLED LDS buffer laid out as
//   [fragid][lane][16 contiguous bf16]   (32B per lane -> 2x ds_load_b128)
__device__ __forceinline__ v16bf lds_load_bfrag(const bf16_t* wbuf, int fragid) {
  const int lane = threadIdx.x & 31;
  const bf16_t* src = wbuf + ((fragid * 32 + lane) << 4);
  v16bf b;
#pragma unroll
  for (int e = 0; e < 16; ++e) b[e] = src[e];
  return b;
}

// Pack 4 floats -> 4 bf16 and store as one 8-byte LDS write.
__device__ __forceinline__ void store_bf16x4(bf16_t* dst, float a, float b,
                                             float c, float d) {
  union { bf16_t h[4]; unsigned long long u; } pk;
  pk.h[0] = (bf16_t)a; pk.h[1] = (bf16_t)b;
  pk.h[2] = (bf16_t)c; pk.h[3] = (bf16_t)d;
  *(unsigned long long*)dst = pk.u;
}

// ---------------------------------------------------------------------------
// Generic tiled GEMM: C[M,N] = act(A[M,K] @ B[K,N] + bias), fp32 in/out,
// bf16 WMMA inner.  M,N multiples of 64; K multiple of 32; block tile 64x64.
// ---------------------------------------------------------------------------
template <int ACT>
__global__ __launch_bounds__(128) void gemm_bias_act(
    const float* __restrict__ A, const float* __restrict__ B,
    const float* __restrict__ bias, float* __restrict__ C,
    int M, int N, int K) {
  __shared__ __align__(16) bf16_t As[64 * 32];   // row-major A tile
  __shared__ __align__(16) bf16_t Bs[4 * 32 * 16]; // 4 swizzled B fragments
  const int tid  = threadIdx.x;
  const int wave = tid >> 5;
  const int lane = tid & 31;
  const int m0 = blockIdx.y * 64;
  const int n0 = blockIdx.x * 64;

  v8f acc[4];
#pragma unroll
  for (int t = 0; t < 4; ++t)
#pragma unroll
    for (int e = 0; e < 8; ++e) acc[t][e] = 0.0f;

  for (int k0 = 0; k0 < K; k0 += 32) {
    // --- stage A 64x32: float4 reads, packed 8B bf16 stores -------------
#pragma unroll
    for (int q = 0; q < 4; ++q) {
      int c  = tid + q * 128;                 // 512 float4 chunks
      int ar = c >> 3, ac = (c & 7) * 4;
      float4 f = *(const float4*)&A[(size_t)(m0 + ar) * K + (k0 + ac)];
      store_bf16x4(&As[ar * 32 + ac], f.x, f.y, f.z, f.w);
    }
    // --- stage B 32x64 directly in B-fragment layout --------------------
    {
      int l  = tid & 31, nt = tid >> 5;       // 128 chunks of 16 elems
      int kk = k0 + (l >> 4) * 16;
      int n  = n0 + nt * 16 + (l & 15);
      bf16_t tmp[16];
#pragma unroll
      for (int e = 0; e < 16; ++e) tmp[e] = (bf16_t)B[(size_t)(kk + e) * N + n];
      bf16_t* dst = Bs + (tid << 4);          // 32B contiguous per thread
#pragma unroll
      for (int e = 0; e < 16; ++e) dst[e] = tmp[e];
    }
    // --- prefetch next K-step tiles (global_prefetch_b8) -----------------
    if (k0 + 32 < K) {
      __builtin_prefetch(&A[(size_t)(m0 + (tid >> 1)) * K + (k0 + 32) + (tid & 1) * 16], 0, 1);
      __builtin_prefetch(&B[(size_t)(k0 + 32 + (tid >> 2)) * N + n0 + (tid & 3) * 16], 0, 1);
    }
    __syncthreads();
    v16bf a = lds_load_a16x32(As + wave * 16 * 32, 32);
#pragma unroll
    for (int t = 0; t < 4; ++t) {
      v16bf b = lds_load_bfrag(Bs, t);
      acc[t] = wmma_bf16(a, b, acc[t]);
    }
    __syncthreads();
  }

  const int mBase = m0 + wave * 16 + 8 * (lane >> 4);
  const int nIn   = lane & 15;
#pragma unroll
  for (int t = 0; t < 4; ++t) {
    int n = n0 + t * 16 + nIn;
    float bv = bias ? bias[n] : 0.0f;
#pragma unroll
    for (int r = 0; r < 8; ++r) {
      float v = acc[t][r] + bv;
      if (ACT) v = v > 0.0f ? v : 0.0f;
      C[(size_t)(mBase + r) * N + n] = v;
    }
  }
}

// ---------------------------------------------------------------------------
// Softmax stats over n values (single block): out[0]=max, out[1]=sum(exp(v-max))
// ---------------------------------------------------------------------------
__global__ __launch_bounds__(1024) void softmax_stats(
    const float* __restrict__ v, int n, float* __restrict__ out) {
  __shared__ float red[1024];
  const int tid = threadIdx.x;
  float m = -3.4e38f;
  for (int i = tid; i < n; i += 1024) m = fmaxf(m, v[i]);
  red[tid] = m; __syncthreads();
  for (int s = 512; s > 0; s >>= 1) {
    if (tid < s) red[tid] = fmaxf(red[tid], red[tid + s]);
    __syncthreads();
  }
  float gmax = red[0]; __syncthreads();
  float su = 0.0f;
  for (int i = tid; i < n; i += 1024) su += __expf(v[i] - gmax);
  red[tid] = su; __syncthreads();
  for (int s = 512; s > 0; s >>= 1) {
    if (tid < s) red[tid] += red[tid + s];
    __syncthreads();
  }
  if (tid == 0) { out[0] = gmax; out[1] = red[0]; }
}

// ---------------------------------------------------------------------------
// Exact k-th largest (single block): bitwise binary search over the monotonic
// float->uint key.  Writes the k-th largest value to out[2].
// ---------------------------------------------------------------------------
__device__ __forceinline__ unsigned f2key(float f) {
  unsigned u = __float_as_uint(f);
  return u ^ ((u >> 31) ? 0xFFFFFFFFu : 0x80000000u);
}

__global__ __launch_bounds__(1024) void kth_largest(
    const float* __restrict__ v, int n, int k, float* __restrict__ out) {
  __shared__ unsigned redc[1024];
  __shared__ unsigned curT;
  const int tid = threadIdx.x;
  if (tid == 0) curT = 0u;
  __syncthreads();
  for (int b = 31; b >= 0; --b) {
    unsigned cand = curT | (1u << b);
    unsigned cnt = 0;
    for (int i = tid; i < n; i += 1024) cnt += (f2key(v[i]) >= cand) ? 1u : 0u;
    redc[tid] = cnt; __syncthreads();
    for (int s = 512; s > 0; s >>= 1) {
      if (tid < s) redc[tid] += redc[tid + s];
      __syncthreads();
    }
    if (tid == 0 && redc[0] >= (unsigned)k) curT = cand;
    __syncthreads();
  }
  if (tid == 0) {
    unsigned T = curT;
    unsigned bits = (T & 0x80000000u) ? (T ^ 0x80000000u) : ~T;
    out[2] = __uint_as_float(bits);
  }
}

// ---------------------------------------------------------------------------
// rX assembly: rows<OLD copy features_o, new rows = sigmoid(ALPHA*(p - tau))
// with p = softmax(slog), tau = softmax prob of the k-th largest logit.
// ---------------------------------------------------------------------------
__global__ __launch_bounds__(256) void build_rx(
    const float* __restrict__ slog, const float* __restrict__ feat_o,
    const float* __restrict__ stats, float* __restrict__ rX,
    float* __restrict__ outRX) {
  int idx = blockIdx.x * 256 + threadIdx.x;
  if (idx >= NTOT * LATD) return;
  int row = idx / LATD;
  float val;
  if (row < OLDN) {
    val = feat_o[idx];
  } else {
    float gmax = stats[0], inv = 1.0f / stats[1];
    float tau = __expf(stats[2] - gmax) * inv;
    float p   = __expf(slog[idx - OLDN * LATD] - gmax) * inv;
    float t   = ALPHA_C * (p - tau);
    val = 1.0f / (1.0f + __expf(-t));
  }
  rX[idx] = val;
  outRX[idx] = val;
}

// ---------------------------------------------------------------------------
// Pair edge-MLP.  Weights staged per layer into LDS *pre-swizzled* into the
// WMMA B-fragment layout: fragid = kb*NT + nt, 16 contiguous bf16 per lane.
// ---------------------------------------------------------------------------
__device__ __forceinline__ void stage_weights_sw(const float* __restrict__ W,
                                                 const float* __restrict__ Bv,
                                                 int K, int N,
                                                 bf16_t* wbuf, float* bbuf) {
  __syncthreads();                               // prev layer done with wbuf
  const int NT = N >> 4;
  const int chunks = (K * N) >> 4;               // 16 elems per chunk
  for (int c = threadIdx.x; c < chunks; c += 256) {
    int l      = c & 31;
    int fragid = c >> 5;
    int nt     = fragid % NT;
    int kb     = fragid / NT;
    int k0     = kb * 32 + (l >> 4) * 16;
    int n      = nt * 16 + (l & 15);
    bf16_t tmp[16];
#pragma unroll
    for (int e = 0; e < 16; ++e) tmp[e] = (bf16_t)W[(k0 + e) * N + n];
    bf16_t* dst = wbuf + (c << 4);               // 32B contiguous
#pragma unroll
    for (int e = 0; e < 16; ++e) dst[e] = tmp[e];
  }
  for (int e = threadIdx.x; e < N; e += 256) bbuf[e] = Bv[e];
  __syncthreads();
}

template <int K, int N>
__device__ __forceinline__ void mlp_layer(bf16_t* a_lds,
                                          const bf16_t* __restrict__ wbuf,
                                          const float* __restrict__ bbuf) {
  const int lane = threadIdx.x & 31;
  constexpr int NT = N / 16, KB = K / 32;
  v8f acc[NT];
#pragma unroll
  for (int t = 0; t < NT; ++t)
#pragma unroll
    for (int e = 0; e < 8; ++e) acc[t][e] = 0.0f;
#pragma unroll
  for (int kb = 0; kb < KB; ++kb) {
    v16bf a = lds_load_a16x32(a_lds + kb * 32, 128);
#pragma unroll
    for (int t = 0; t < NT; ++t) {
      v16bf b = lds_load_bfrag(wbuf, kb * NT + t);
      acc[t] = wmma_bf16(a, b, acc[t]);
    }
  }
  const int mOff = 8 * (lane >> 4);
  const int nIn  = lane & 15;
  // In-place update is safe: LDS ops from one wave execute in order and the
  // region is wave-private; stores below are issued after all loads above.
#pragma unroll
  for (int t = 0; t < NT; ++t) {
    float bv = bbuf[t * 16 + nIn];
#pragma unroll
    for (int r = 0; r < 8; ++r) {
      float v = acc[t][r] + bv;
      v = v > 0.0f ? v : 0.0f;
      a_lds[(r + mOff) * 128 + t * 16 + nIn] = (bf16_t)v;
    }
  }
}

__global__ __launch_bounds__(256) void pair_mlp(
    const float* __restrict__ rX,
    const float* __restrict__ w11, const float* __restrict__ b11,
    const float* __restrict__ w12, const float* __restrict__ b12,
    const float* __restrict__ w13, const float* __restrict__ b13,
    const float* __restrict__ w14, const float* __restrict__ b14,
    const float* __restrict__ w15, const float* __restrict__ b15,
    const float* __restrict__ w16, const float* __restrict__ b16,
    const int* __restrict__ mask, const int* __restrict__ mint,
    float* __restrict__ rAlog) {
  __shared__ __align__(16) bf16_t wbuf[128 * 64];     // 16 KB (max layer)
  __shared__ __align__(16) bf16_t act[8][16 * 128];   // 32 KB per-wave tiles
  __shared__ float bbuf[128];
  __shared__ float rxn[128];

  const int tid  = threadIdx.x;
  const int wave = tid >> 5;
  const int lane = tid & 31;
  const int i  = blockIdx.y;             // new node 0..255
  const int jw = blockIdx.x * 128 + wave * 16;

  if (tid < 128) rxn[tid] = rX[(size_t)(OLDN + i) * LATD + tid];
  __syncthreads();

  {  // build s tile (16x128) = rX[jw+r] * rxn : float4 reads, 8B bf16 stores
    bf16_t* a = act[wave];
#pragma unroll
    for (int q = 0; q < 16; ++q) {
      int c4 = lane + q * 32;            // 512 chunks of 4 cols
      int r  = c4 >> 5, c = (c4 & 31) * 4;
      float4 f  = *(const float4*)&rX[(size_t)(jw + r) * LATD + c];
      float4 rm = *(const float4*)&rxn[c];
      store_bf16x4(&a[r * 128 + c], f.x * rm.x, f.y * rm.y, f.z * rm.z, f.w * rm.w);
    }
  }

  stage_weights_sw(w11, b11, 128, 64, wbuf, bbuf);
  mlp_layer<128, 64>(act[wave], wbuf, bbuf);
  stage_weights_sw(w12, b12, 64, 128, wbuf, bbuf);
  mlp_layer<64, 128>(act[wave], wbuf, bbuf);
  stage_weights_sw(w13, b13, 128, 64, wbuf, bbuf);
  mlp_layer<128, 64>(act[wave], wbuf, bbuf);
  stage_weights_sw(w14, b14, 64, 64, wbuf, bbuf);
  mlp_layer<64, 64>(act[wave], wbuf, bbuf);

  // ---- layer 5 (64->32): keep D in registers, fold layer 6 (32->1) via shfl
  stage_weights_sw(w15, b15, 64, 32, wbuf, bbuf);
  v8f acc5[2];
#pragma unroll
  for (int t = 0; t < 2; ++t)
#pragma unroll
    for (int e = 0; e < 8; ++e) acc5[t][e] = 0.0f;
#pragma unroll
  for (int kb = 0; kb < 2; ++kb) {
    v16bf a = lds_load_a16x32(act[wave] + kb * 32, 128);
#pragma unroll
    for (int t = 0; t < 2; ++t) {
      v16bf b = lds_load_bfrag(wbuf, kb * 2 + t);
      acc5[t] = wmma_bf16(a, b, acc5[t]);
    }
  }
  const int hi  = lane >> 4;
  const int nIn = lane & 15;
  const float w16a = w16[nIn],  w16b = w16[16 + nIn];
  const float b5a  = bbuf[nIn], b5b  = bbuf[16 + nIn];
  float part[8];
#pragma unroll
  for (int r = 0; r < 8; ++r) {
    float v0 = acc5[0][r] + b5a; v0 = v0 > 0.0f ? v0 : 0.0f;
    float v1 = acc5[1][r] + b5b; v1 = v1 > 0.0f ? v1 : 0.0f;
    part[r] = v0 * w16a + v1 * w16b;
  }
#pragma unroll
  for (int m = 1; m < 16; m <<= 1)
#pragma unroll
    for (int r = 0; r < 8; ++r) part[r] += __shfl_xor(part[r], m, 32);
  if (nIn == 0) {
    const float bb = b16[0];
#pragma unroll
    for (int r = 0; r < 8; ++r) {
      int j = jw + r + 8 * hi;
      size_t mi = (size_t)i * NTOT + j;
      bool keep = (mask[mi] != 0) && (mint[mi] == 0);
      rAlog[mi] = keep ? (part[r] + bb) : NEGINF;
    }
  }
}

// ---------------------------------------------------------------------------
// rA finalization: soft-top-k of softmax probs, internal edges forced to 1.
// ---------------------------------------------------------------------------
__global__ __launch_bounds__(256) void build_ra_final(
    const float* __restrict__ rAlog, const int* __restrict__ mint,
    const float* __restrict__ stats, float* __restrict__ rAfin) {
  int idx = blockIdx.x * 256 + threadIdx.x;
  if (idx >= NNEW * NTOT) return;
  if (mint[idx] != 0) { rAfin[idx] = 1.0f; return; }
  float gmax = stats[0], inv = 1.0f / stats[1];
  float tau = __expf(stats[2] - gmax) * inv;
  float p   = __expf(rAlog[idx] - gmax) * inv;
  float t   = ALPHA_C * (p - tau);
  rAfin[idx] = 1.0f / (1.0f + __expf(-t));
}

// ---------------------------------------------------------------------------
// A = tril(Afull,-1) + tril(Afull,-1)^T + I, with Afull = [[adj_o, 0],[rAfin]]
// ---------------------------------------------------------------------------
__global__ __launch_bounds__(256) void assemble_A(
    const float* __restrict__ adj_o, const float* __restrict__ rAfin,
    float* __restrict__ out) {
  size_t idx = (size_t)blockIdx.x * 256 + threadIdx.x;
  int i = (int)(idx >> 12);
  int j = (int)(idx & 4095);
  float v;
  if (i == j) {
    v = 1.0f;
  } else {
    int r = i > j ? i : j;
    int c = i > j ? j : i;
    v = (r < OLDN) ? adj_o[(size_t)r * OLDN + c]
                   : rAfin[(size_t)(r - OLDN) * NTOT + c];
  }
  out[idx] = v;
}

// ---------------------------------------------------------------------------
// Host-side orchestration
// ---------------------------------------------------------------------------
extern "C" void kernel_launch(void* const* d_in, const int* in_sizes, int n_in,
                              void* d_out, int out_size, void* d_ws, size_t ws_size,
                              hipStream_t stream) {
  const float* x     = (const float*)d_in[0];
  const float* adj   = (const float*)d_in[1];
  const int*   mask  = (const int*)d_in[2];
  const float* adj_o = (const float*)d_in[3];
  const float* feat  = (const float*)d_in[4];
  const int*   mint  = (const int*)d_in[5];
  const float* gc1w  = (const float*)d_in[6];
  const float* gc1b  = (const float*)d_in[7];
  const float* gc2w  = (const float*)d_in[8];
  const float* gc2b  = (const float*)d_in[9];
  const float* w11   = (const float*)d_in[10];
  const float* b11   = (const float*)d_in[11];
  const float* w12   = (const float*)d_in[12];
  const float* b12   = (const float*)d_in[13];
  const float* w13   = (const float*)d_in[14];
  const float* b13   = (const float*)d_in[15];
  const float* w14   = (const float*)d_in[16];
  const float* b14   = (const float*)d_in[17];
  const float* w15   = (const float*)d_in[18];
  const float* b15   = (const float*)d_in[19];
  const float* w16   = (const float*)d_in[20];
  const float* b16   = (const float*)d_in[21];
  const float* w21   = (const float*)d_in[22];
  const float* b21   = (const float*)d_in[23];
  const float* w22   = (const float*)d_in[24];
  const float* b22   = (const float*)d_in[25];
  const float* w23   = (const float*)d_in[26];
  const float* b23   = (const float*)d_in[27];

  float* ws = (float*)d_ws;
  size_t off = 0;
  float* xw     = ws + off; off += (size_t)NTOT * HIDD;   // 4096x256
  float* h      = ws + off; off += (size_t)NTOT * HIDD;   // 4096x256
  float* hw     = ws + off; off += (size_t)NTOT * LATD;   // 4096x128
  float* zn     = ws + off; off += (size_t)NNEW * LATD;   // 256x128
  float* r1     = ws + off; off += (size_t)NNEW * R1D;    // 256x256
  float* r2     = ws + off; off += (size_t)NNEW * R2D;    // 256x256
  float* slog   = ws + off; off += (size_t)NNEW * LATD;   // 256x128
  float* statsX = ws + off; off += 8;
  float* rXf    = ws + off; off += (size_t)NTOT * LATD;   // 4096x128
  float* rAlog  = ws + off; off += (size_t)NNEW * NTOT;   // 256x4096
  float* statsA = ws + off; off += 8;
  float* rAfin  = ws + off; off += (size_t)NNEW * NTOT;   // 256x4096

  float* outA  = (float*)d_out;
  float* outRX = (float*)d_out + (size_t)NTOT * NTOT;

  // GCN: h = relu(adj @ (x @ gc1_w) + b1); zn = relu(adj[OLD:] @ (h @ gc2_w) + b2)
  gemm_bias_act<0><<<dim3(HIDD / 64, NTOT / 64), 128, 0, stream>>>(
      x, gc1w, nullptr, xw, NTOT, HIDD, LATD);
  gemm_bias_act<1><<<dim3(HIDD / 64, NTOT / 64), 128, 0, stream>>>(
      adj, xw, gc1b, h, NTOT, HIDD, NTOT);
  gemm_bias_act<0><<<dim3(LATD / 64, NTOT / 64), 128, 0, stream>>>(
      h, gc2w, nullptr, hw, NTOT, LATD, HIDD);
  gemm_bias_act<1><<<dim3(LATD / 64, NNEW / 64), 128, 0, stream>>>(
      adj + (size_t)OLDN * NTOT, hw, gc2b, zn, NNEW, LATD, NTOT);

  // Feature MLP: 128 -> 256 -> 256 -> 128 (last no act)
  gemm_bias_act<1><<<dim3(R1D / 64, NNEW / 64), 128, 0, stream>>>(
      zn, w21, b21, r1, NNEW, R1D, LATD);
  gemm_bias_act<1><<<dim3(R2D / 64, NNEW / 64), 128, 0, stream>>>(
      r1, w22, b22, r2, NNEW, R2D, R1D);
  gemm_bias_act<0><<<dim3(LATD / 64, NNEW / 64), 128, 0, stream>>>(
      r2, w23, b23, slog, NNEW, LATD, R2D);

  // rX soft-top-k (softmax + exact 1024-th largest threshold)
  softmax_stats<<<1, 1024, 0, stream>>>(slog, NNEW * LATD, statsX);
  kth_largest<<<1, 1024, 0, stream>>>(slog, NNEW * LATD, BUDGET_X, statsX);
  build_rx<<<(NTOT * LATD + 255) / 256, 256, 0, stream>>>(
      slog, feat, statsX, rXf, outRX);

  // Pair edge-MLP -> masked logits (dominant compute, all WMMA)
  pair_mlp<<<dim3(NTOT / 128, NNEW), 256, 0, stream>>>(
      rXf, w11, b11, w12, b12, w13, b13, w14, b14, w15, b15, w16, b16,
      mask, mint, rAlog);

  // rA soft-top-k (softmax over 1M + exact 5000-th largest)
  softmax_stats<<<1, 1024, 0, stream>>>(rAlog, NNEW * NTOT, statsA);
  kth_largest<<<1, 1024, 0, stream>>>(rAlog, NNEW * NTOT, BUDGET_A, statsA);
  build_ra_final<<<(NNEW * NTOT + 255) / 256, 256, 0, stream>>>(
      rAlog, mint, statsA, rAfin);

  // Final symmetric adjacency
  assemble_A<<<(int)(((size_t)NTOT * NTOT) / 256), 256, 0, stream>>>(
      adj_o, rAfin, outA);
}
